// SelfAttention_13632226198099
// MI455X (gfx1250) — matrix operand order repaired and checked by reading, not compiled, not run on hardware
//
#include <hip/hip_runtime.h>

// ---------------- problem constants ----------------
#define BB 2
#define SS 2048
#define EE 1024
#define HH 16
#define DD 64          // EE/HH

typedef unsigned short ushort_t;
typedef __attribute__((ext_vector_type(16))) __bf16    v16bf;
typedef __attribute__((ext_vector_type(8)))  float     v8f;
typedef __attribute__((ext_vector_type(8)))  ushort_t  v8u;
typedef __attribute__((ext_vector_type(16))) ushort_t  v16u;
typedef __attribute__((ext_vector_type(4)))  int       v4i;

#if __has_builtin(__builtin_amdgcn_global_load_async_to_lds_b128) && \
    __has_builtin(__builtin_amdgcn_s_wait_asynccnt)
#define USE_ASYNC 1
#else
#define USE_ASYNC 0
#endif

// fp32 -> bf16 round-to-nearest-even
__device__ __forceinline__ ushort_t f2bf(float f) {
  unsigned int u = __float_as_uint(f);
  u += 0x7FFFu + ((u >> 16) & 1u);
  return (ushort_t)(u >> 16);
}

__device__ __forceinline__ v8f wmma_bf16(v16bf a, v16bf b, v8f c) {
  return __builtin_amdgcn_wmma_f32_16x16x32_bf16(false, a, false, b, (short)0, c,
                                                 false, false);
}

// A-matrix fragment (16x32 bf16, ISA 7.12.2): lane holds row `row`;
// element i -> K offset:  i<8 ? half*8+i : 16+half*8+(i-8)
__device__ __forceinline__ v16bf load_a_frag(const ushort_t* base, int lda,
                                             int row, int kbase, int half) {
  const ushort_t* p = base + (size_t)row * lda + kbase + half * 8;
  v8u lo = *(const v8u*)p;
  v8u hi = *(const v8u*)(p + 16);
  v16u r;
#pragma unroll
  for (int i = 0; i < 8; ++i) { r[i] = lo[i]; r[i + 8] = hi[i]; }
  return __builtin_bit_cast(v16bf, r);
}

// B-matrix fragment (32x16 bf16): krow = kbase+lane; lane holds column
// nbase+lane%16, element i -> K offset: i<8 ? half*8+i : 16+half*8+(i-8)
__device__ __forceinline__ v16bf load_b_frag(const ushort_t* base, size_t ldb,
                                             int krow, int nbase) {
  const int kb = krow & ~31, l = krow & 31, hf = l >> 4;
  const ushort_t* p = base + (size_t)kb * ldb + nbase + (l & 15);
  v16u r;
#pragma unroll
  for (int i = 0; i < 8; ++i) {
    r[i] = p[(size_t)(hf * 8 + i) * ldb];
    r[i + 8] = p[(size_t)(16 + hf * 8 + i) * ldb];
  }
  return __builtin_bit_cast(v16bf, r);
}

// 16-byte tile-staging copy: global -> LDS (async DMA when available)
__device__ __forceinline__ void stage16(const ushort_t* gsrc, ushort_t* ldst) {
#if USE_ASYNC
  typedef __attribute__((address_space(1))) v4i* gp_t;
  typedef __attribute__((address_space(3))) v4i* lp_t;
  __builtin_amdgcn_global_load_async_to_lds_b128(
      (gp_t)(const_cast<ushort_t*>(gsrc)), (lp_t)ldst, 0, 0);
#else
  *(v8u*)ldst = *(const v8u*)gsrc;
#endif
}

__device__ __forceinline__ void stage_wait_tail16() {
#if USE_ASYNC
  __builtin_amdgcn_s_wait_asynccnt(16);  // current tile landed; next 16 in flight
#endif
  asm volatile("" ::: "memory");
}
__device__ __forceinline__ void stage_wait_all() {
#if USE_ASYNC
  __builtin_amdgcn_s_wait_asynccnt(0);
#endif
  asm volatile("" ::: "memory");
}

// ---------------- kernel 1: fp32 -> bf16 ----------------
__global__ __launch_bounds__(256) void cvt_kernel(const float* __restrict__ in,
                                                  ushort_t* __restrict__ out, int n) {
  int i = blockIdx.x * 256 + threadIdx.x;
  if (i < n) out[i] = f2bf(in[i]);
}

// ---------------- kernel 2: W (E,E) -> W^T bf16 ----------------
__global__ __launch_bounds__(256) void wtr_kernel(const float* __restrict__ W,
                                                  ushort_t* __restrict__ WT) {
  int i = blockIdx.x * 256 + threadIdx.x;   // i = k*E + n
  int k = i >> 10, n = i & (EE - 1);
  WT[i] = f2bf(W[(size_t)n * EE + k]);
}

// ---------------- kernel 3: fused QKV projection (32x64 tile / wave) ----------------
// blockIdx.x = mode*(128*16) + mtile*16 + head ; mode 0=Q,1=K,2=V
__global__ __launch_bounds__(32) void proj_qkv_kernel(
    const ushort_t* __restrict__ X,    // (B*S, E) bf16
    const ushort_t* __restrict__ WTq, const ushort_t* __restrict__ WTk,
    const ushort_t* __restrict__ WTv,
    ushort_t* __restrict__ Qo,         // (B,H,S,D)
    ushort_t* __restrict__ KTo,        // (B,H,D,S)
    ushort_t* __restrict__ Vo) {       // (B,H,S,D)
  const int lane = threadIdx.x;
  const int m = lane & 15, half = lane >> 4;
  const int bid  = blockIdx.x;
  const int mode = bid / (128 * 16);
  const int rem  = bid % (128 * 16);
  const int mt   = rem >> 4;          // 32-token tile
  const int h    = rem & 15;
  const ushort_t* WT = (mode == 0) ? WTq : (mode == 1) ? WTk : WTv;
  const int nbase = h * DD;

  v8f acc[2][4];
#pragma unroll
  for (int u = 0; u < 2; ++u)
#pragma unroll
    for (int f = 0; f < 4; ++f)
#pragma unroll
      for (int r = 0; r < 8; ++r) acc[u][f][r] = 0.f;

#pragma unroll 2
  for (int c = 0; c < EE / 32; ++c) {
    v16bf a0 = load_a_frag(X, EE, mt * 32 + m, c * 32, half);
    v16bf a1 = load_a_frag(X, EE, mt * 32 + 16 + m, c * 32, half);
#pragma unroll
    for (int f = 0; f < 4; ++f) {
      v16bf b = load_b_frag(WT, EE, c * 32 + lane, nbase + f * 16);
      acc[0][f] = wmma_bf16(a0, b, acc[0][f]);
      acc[1][f] = wmma_bf16(a1, b, acc[1][f]);
    }
  }

#pragma unroll
  for (int u = 0; u < 2; ++u)
#pragma unroll
    for (int f = 0; f < 4; ++f)
#pragma unroll
      for (int r = 0; r < 8; ++r) {
        int tok = mt * 32 + u * 16 + r + 8 * half;
        int b_  = tok / SS, s_ = tok % SS;
        int d   = f * 16 + m;
        ushort_t v = f2bf(acc[u][f][r]);
        if (mode == 1)
          KTo[(((size_t)(b_ * HH + h)) * DD + d) * SS + s_] = v;
        else if (mode == 0)
          Qo[(((size_t)(b_ * HH + h)) * SS + s_) * DD + d] = v;
        else
          Vo[(((size_t)(b_ * HH + h)) * SS + s_) * DD + d] = v;
      }
}

// ---------------- kernel 4: flash attention ----------------
// blockIdx.x = ((b*H + h) * (S/16)) + qtile ; one wave handles 16 query rows.
// K/V tiles staged in LDS via double-buffered async copies.
__global__ __launch_bounds__(32) void attn_kernel(
    const ushort_t* __restrict__ Q,   // (B,H,S,D)
    const ushort_t* __restrict__ KT,  // (B,H,D,S)
    const ushort_t* __restrict__ V,   // (B,H,S,D)
    const unsigned char* __restrict__ mask,  // (B,S)
    ushort_t* __restrict__ O) {       // (B,S,E) bf16 (pre-projection)
  const int lane = threadIdx.x;
  const int m = lane & 15, half = lane >> 4;
  const int qt = blockIdx.x % (SS / 16);
  const int h  = (blockIdx.x / (SS / 16)) % HH;
  const int b  = blockIdx.x / ((SS / 16) * HH);

  const ushort_t* Qh  = Q  + ((size_t)(b * HH + h)) * SS * DD;
  const ushort_t* KTh = KT + ((size_t)(b * HH + h)) * DD * SS;
  const ushort_t* Vh  = V  + ((size_t)(b * HH + h)) * SS * DD;
  const unsigned char* mk = mask + (size_t)b * SS;
  const float sc = 0.125f;  // 1/sqrt(D)

  __shared__ __align__(16) ushort_t Kbuf[2][64 * 32];  // (d, key_local)
  __shared__ __align__(16) ushort_t Vbuf[2][32 * 64];  // (key_local, d)
  __shared__ __align__(16) ushort_t Pld[16 * 32];

  // Q A-fragments (d-chunks of 32), resident for the whole loop
  v16bf aq[2];
#pragma unroll
  for (int c = 0; c < 2; ++c) aq[c] = load_a_frag(Qh, DD, qt * 16 + m, c * 32, half);

  v8f acc[4];
  float rm[8], rl[8];
#pragma unroll
  for (int r = 0; r < 8; ++r) { rm[r] = -1.0e30f; rl[r] = 0.f; }
#pragma unroll
  for (int f = 0; f < 4; ++f)
#pragma unroll
    for (int r = 0; r < 8; ++r) acc[f][r] = 0.f;

  // stage one 32-key tile (K: 64x32, V: 32x64): 8+8 b128 transfers per lane
  auto stage_tile = [&](int kb, int buf) {
#pragma unroll
    for (int j = 0; j < 8; ++j) {            // K^T tile: 256 x 16B
      int lin = j * 32 + lane;
      int row = lin >> 2, ch = lin & 3;      // row: d 0..63, ch: 4 x 8 elems
      stage16(KTh + (size_t)row * SS + kb + ch * 8, &Kbuf[buf][row * 32 + ch * 8]);
    }
#pragma unroll
    for (int j = 0; j < 8; ++j) {            // V tile: 256 x 16B
      int lin = j * 32 + lane;
      int row = lin >> 3, ch = lin & 7;      // row: key 0..31, ch: 8 x 8 elems
      stage16(Vh + (size_t)(kb + row) * DD + ch * 8, &Vbuf[buf][row * 64 + ch * 8]);
    }
  };

  stage_tile(0, 0);
  int cur = 0;
  for (int kb = 0; kb < SS; kb += 32) {
    if (kb + 32 < SS) {            // keep next tile's 16 transfers in flight
      stage_tile(kb + 32, cur ^ 1);
      stage_wait_tail16();
    } else {
      stage_wait_all();
    }
    const ushort_t* Kb = Kbuf[cur];
    const ushort_t* Vb = Vbuf[cur];

    // ---- scores: two 16x16 tiles over 32 keys ----
    v8f s0, s1;
#pragma unroll
    for (int r = 0; r < 8; ++r) { s0[r] = 0.f; s1[r] = 0.f; }
#pragma unroll
    for (int c = 0; c < 2; ++c) {
      v16bf bk0 = load_b_frag(Kb, 32, c * 32 + lane, 0);
      v16bf bk1 = load_b_frag(Kb, 32, c * 32 + lane, 16);
      s0 = wmma_bf16(aq[c], bk0, s0);
      s1 = wmma_bf16(aq[c], bk1, s1);
    }
    const float ninf = -__builtin_inff();
    float msc0 = mk[kb + m]      ? 0.f : ninf;   // this lane's key column, tile0
    float msc1 = mk[kb + 16 + m] ? 0.f : ninf;   // tile1

    float f0[8], f1[8];
#pragma unroll
    for (int r = 0; r < 8; ++r) {
      f0[r] = s0[r] * sc + msc0;
      f1[r] = s1[r] * sc + msc1;
    }
    // ---- online softmax (row = r + 8*half across one 16-lane half) ----
#pragma unroll
    for (int r = 0; r < 8; ++r) {
      float t = fmaxf(f0[r], f1[r]);
      t = fmaxf(t, __shfl_xor(t, 1, 32));
      t = fmaxf(t, __shfl_xor(t, 2, 32));
      t = fmaxf(t, __shfl_xor(t, 4, 32));
      t = fmaxf(t, __shfl_xor(t, 8, 32));
      float nmax = fmaxf(rm[r], t);
      float corr = __expf(rm[r] - nmax);
      rm[r] = nmax;
      float p0 = __expf(f0[r] - nmax);
      float p1 = __expf(f1[r] - nmax);
      float srow = p0 + p1;
      srow += __shfl_xor(srow, 1, 32);
      srow += __shfl_xor(srow, 2, 32);
      srow += __shfl_xor(srow, 4, 32);
      srow += __shfl_xor(srow, 8, 32);
      rl[r] = rl[r] * corr + srow;
#pragma unroll
      for (int f = 0; f < 4; ++f) acc[f][r] *= corr;
      f0[r] = p0; f1[r] = p1;
    }
    // ---- reshape P: C-layout -> LDS -> A-layout ----
#pragma unroll
    for (int r = 0; r < 8; ++r) {
      int row = r + 8 * half;
      Pld[row * 32 + m]      = f2bf(f0[r]);
      Pld[row * 32 + 16 + m] = f2bf(f1[r]);
    }
    v16bf ap = load_a_frag(Pld, 32, m, 0, half);
    // ---- O += P @ V ----
#pragma unroll
    for (int f = 0; f < 4; ++f) {
      v16bf bv = load_b_frag(Vb, 64, lane, f * 16);
      acc[f] = wmma_bf16(ap, bv, acc[f]);
    }
    cur ^= 1;
  }

  // ---- epilogue: normalize, write O in (b, s, h*64+d) ----
#pragma unroll
  for (int r = 0; r < 8; ++r) {
    float inv = (rl[r] > 0.f) ? (1.f / rl[r]) : 0.f;
    int s_ = qt * 16 + r + 8 * half;
#pragma unroll
    for (int f = 0; f < 4; ++f) {
      int col = h * DD + f * 16 + m;
      O[((size_t)b * SS + s_) * EE + col] = f2bf(acc[f][r] * inv);
    }
  }
}

// ---------------- kernel 5: output projection (32x64 tile / wave, fp32 out) ----------------
// blockIdx.x = mtile*16 + ntile  (ntile spans 64 cols)
__global__ __launch_bounds__(32) void out_proj_kernel(
    const ushort_t* __restrict__ Ob,  // (B*S, E) bf16
    const ushort_t* __restrict__ WTo, // (E, E) bf16, K-major
    float* __restrict__ out) {        // (B*S, E) fp32
  const int lane = threadIdx.x;
  const int m = lane & 15, half = lane >> 4;
  const int mt = blockIdx.x >> 4;     // 32-token tile
  const int nt = blockIdx.x & 15;
  const int nbase = nt * 64;

  v8f acc[2][4];
#pragma unroll
  for (int u = 0; u < 2; ++u)
#pragma unroll
    for (int f = 0; f < 4; ++f)
#pragma unroll
      for (int r = 0; r < 8; ++r) acc[u][f][r] = 0.f;

#pragma unroll 2
  for (int c = 0; c < EE / 32; ++c) {
    v16bf a0 = load_a_frag(Ob, EE, mt * 32 + m, c * 32, half);
    v16bf a1 = load_a_frag(Ob, EE, mt * 32 + 16 + m, c * 32, half);
#pragma unroll
    for (int f = 0; f < 4; ++f) {
      v16bf b = load_b_frag(WTo, EE, c * 32 + lane, nbase + f * 16);
      acc[0][f] = wmma_bf16(a0, b, acc[0][f]);
      acc[1][f] = wmma_bf16(a1, b, acc[1][f]);
    }
  }
#pragma unroll
  for (int u = 0; u < 2; ++u)
#pragma unroll
    for (int f = 0; f < 4; ++f)
#pragma unroll
      for (int r = 0; r < 8; ++r) {
        int tok = mt * 32 + u * 16 + r + 8 * half;
        out[(size_t)tok * EE + nbase + f * 16 + m] = acc[u][f][r];
      }
}

// ---------------- host: launch sequence ----------------
extern "C" void kernel_launch(void* const* d_in, const int* in_sizes, int n_in,
                              void* d_out, int out_size, void* d_ws, size_t ws_size,
                              hipStream_t stream) {
  const float* x  = (const float*)d_in[0];
  const unsigned char* msk = (const unsigned char*)d_in[1];
  const float* Wq = (const float*)d_in[2];
  const float* Wk = (const float*)d_in[3];
  const float* Wv = (const float*)d_in[4];
  const float* Wo = (const float*)d_in[5];
  float* out = (float*)d_out;

  char* w = (char*)d_ws;
  const size_t NX = (size_t)BB * SS * EE;      // 4 Mi elems
  const size_t NW = (size_t)EE * EE;           // 1 Mi elems
  const size_t NH = (size_t)BB * HH * SS * DD; // 4 Mi elems
  ushort_t* xbf = (ushort_t*)w;                        w += NX * 2;
  ushort_t* WTq = (ushort_t*)w;                        w += NW * 2;
  ushort_t* WTk = (ushort_t*)w;                        w += NW * 2;
  ushort_t* WTv = (ushort_t*)w;                        w += NW * 2;
  ushort_t* WTo = (ushort_t*)w;                        w += NW * 2;
  ushort_t* Qb  = (ushort_t*)w;                        w += NH * 2;
  ushort_t* KTb = (ushort_t*)w;                        w += NH * 2;
  ushort_t* Vb  = (ushort_t*)w;                        w += NH * 2;
  ushort_t* Ob  = (ushort_t*)w;                        w += NX * 2;

  // 1) x -> bf16
  cvt_kernel<<<(int)(NX / 256), 256, 0, stream>>>(x, xbf, (int)NX);
  // 2) weights -> bf16, transposed (K-major)
  wtr_kernel<<<(int)(NW / 256), 256, 0, stream>>>(Wq, WTq);
  wtr_kernel<<<(int)(NW / 256), 256, 0, stream>>>(Wk, WTk);
  wtr_kernel<<<(int)(NW / 256), 256, 0, stream>>>(Wv, WTv);
  wtr_kernel<<<(int)(NW / 256), 256, 0, stream>>>(Wo, WTo);
  // 3) fused QKV projection: 3 modes x 128 M-tiles x 16 heads
  proj_qkv_kernel<<<3 * 128 * 16, 32, 0, stream>>>(xbf, WTq, WTk, WTv, Qb, KTb, Vb);
  // 4) flash attention: B*H*(S/16) waves
  attn_kernel<<<BB * HH * (SS / 16), 32, 0, stream>>>(Qb, KTb, Vb, msk, Ob);
  // 5) output projection
  out_proj_kernel<<<128 * 16, 32, 0, stream>>>(Ob, WTo, out);
}
